// MultiHeadAttentionQuantum_65481071404803
// MI455X (gfx1250) — matrix operand order repaired and checked by reading, not compiled, not run on hardware
//
#include <hip/hip_runtime.h>
#include <hip/hip_bf16.h>

// Quantum multi-head "attention": per-token 4-qubit circuit per head.
// Closed form: out = P_all @ G^T + b, with P from two 16x16 GEMMs per head.
// All GEMMs on v_wmma_f32_16x16x32_f16 (wave32, gfx1250).

#define EMBED 32
#define TOKENS (32 * 4096)
#define TILES (TOKENS / 16)          // 8192 tiles of 16 tokens
#define BLOCKS 256
#define WAVES_PER_BLOCK 8
#define THREADS (WAVES_PER_BLOCK * 32)
#define TILES_PER_WAVE (TILES / (BLOCKS * WAVES_PER_BLOCK))  // 4

typedef __attribute__((ext_vector_type(16))) _Float16 v16h;
typedef __attribute__((ext_vector_type(8)))  _Float16 v8h;
typedef __attribute__((ext_vector_type(8)))  float    v8f;

__device__ __forceinline__ v16h ldpair(const _Float16* p, int gapHalves) {
  v8h lo = *(const v8h*)p;
  v8h hi = *(const v8h*)(p + gapHalves);
  return __builtin_shufflevector(lo, hi, 0,1,2,3,4,5,6,7,8,9,10,11,12,13,14,15);
}

// ---------------- setup: build U_h (16x16 complex, f16, K padded to 32) and G (32x128 f16)
__global__ __launch_bounds__(128) void qmha_setup(const float* __restrict__ params,
                                                  const float* __restrict__ W_out,
                                                  _Float16* __restrict__ Uw,
                                                  _Float16* __restrict__ Gw) {
  const int tid = threadIdx.x;
  const int h = tid >> 4;          // head
  const int j = tid & 15;          // basis-state column
  float sr[16], si[16];
#pragma unroll
  for (int k = 0; k < 16; ++k) { sr[k] = 0.f; si[k] = 0.f; }
  sr[j] = 1.f;
  const float* ph = params + h * 24;   // (LAYERS=2, N_QUBITS=4, 3)
  for (int l = 0; l < 2; ++l) {
    for (int q = 0; q < 4; ++q) {
      const int m = 1 << (3 - q);       // qubit q -> bit (3-q)
      float cx, sx, cy, sy, cz, sz;
      __sincosf(0.5f * ph[(l*4+q)*3+0], &sx, &cx);
      __sincosf(0.5f * ph[(l*4+q)*3+1], &sy, &cy);
      __sincosf(0.5f * ph[(l*4+q)*3+2], &sz, &cz);
#pragma unroll
      for (int k = 0; k < 16; ++k) {
        if (k & m) continue;
        const int k1 = k | m;
        // RX: a0' = c a0 - i s a1 ; a1' = -i s a0 + c a1
        float r0 = sr[k], i0 = si[k], r1 = sr[k1], i1 = si[k1];
        float nr0 = cx*r0 + sx*i1, ni0 = cx*i0 - sx*r1;
        float nr1 = cx*r1 + sx*i0, ni1 = cx*i1 - sx*r0;
        // RY: a0' = c a0 - s a1 ; a1' = s a0 + c a1
        r0 = nr0; i0 = ni0; r1 = nr1; i1 = ni1;
        nr0 = cy*r0 - sy*r1; ni0 = cy*i0 - sy*i1;
        nr1 = sy*r0 + cy*r1; ni1 = sy*i0 + cy*i1;
        // RZ: a0' *= e^{-i t/2} ; a1' *= e^{+i t/2}
        r0 = nr0; i0 = ni0; r1 = nr1; i1 = ni1;
        sr[k]  = cz*r0 + sz*i0;  si[k]  = cz*i0 - sz*r0;
        sr[k1] = cz*r1 - sz*i1;  si[k1] = cz*i1 + sz*r1;
      }
    }
    // CNOT ring: (0,1)(1,2)(2,3)(3,0)
#pragma unroll
    for (int c = 0; c < 4; ++c) {
      const int tq = (c + 1) & 3;
      const int mc = 1 << (3 - c), mt = 1 << (3 - tq);
#pragma unroll
      for (int k = 0; k < 16; ++k) {
        if ((k & mc) && !(k & mt)) {
          const int k1 = k | mt;
          float t0 = sr[k]; sr[k] = sr[k1]; sr[k1] = t0;
          float t1 = si[k]; si[k] = si[k1]; si[k1] = t1;
        }
      }
    }
  }
  // column j of U_h, layout [h][re/im][row n][K=32 padded]
#pragma unroll
  for (int n = 0; n < 16; ++n) {
    Uw[((h*2+0)*16 + n)*32 + j]      = (_Float16)sr[n];
    Uw[((h*2+1)*16 + n)*32 + j]      = (_Float16)si[n];
    Uw[((h*2+0)*16 + n)*32 + 16 + j] = (_Float16)0.f;
    Uw[((h*2+1)*16 + n)*32 + 16 + j] = (_Float16)0.f;
  }
  // G[e][16h+n] = sum_q sign(n,q) * W_out[e][4h+q]
  const int e = tid & 31;
  const int kb = (tid >> 5) * 32;
  for (int k = kb; k < kb + 32; ++k) {
    const int hh = k >> 4, n = k & 15;
    float v = 0.f;
#pragma unroll
    for (int q = 0; q < 4; ++q) {
      const float w = W_out[e*32 + hh*4 + q];
      v += ((n >> (3 - q)) & 1) ? -w : w;
    }
    Gw[e*128 + k] = (_Float16)v;
  }
}

// ---------------- main kernel: 16-token tiles, 24 WMMAs per tile per wave
__global__ __launch_bounds__(THREADS) void qmha_main(const float* __restrict__ x,
                                                     const float* __restrict__ b_out,
                                                     const _Float16* __restrict__ Uw,
                                                     const _Float16* __restrict__ Gw,
                                                     float* __restrict__ out) {
  __shared__ __attribute__((aligned(16))) _Float16 ldsU[8*2*16*32];                 // 16 KB
  __shared__ __attribute__((aligned(16))) _Float16 ldsG[32*128];                    // 8 KB
  __shared__ __attribute__((aligned(16))) _Float16 ldsS0[WAVES_PER_BLOCK][8*16*32]; // 8 KB/wave

  const int tid  = threadIdx.x;
  const int lane = tid & 31;
  const int wave = tid >> 5;
  const int n16  = lane & 15;
  const int hi16 = lane >> 4;

  // cooperative stage of constants into LDS
#pragma unroll
  for (int r = 0; r < 4; ++r)
    ((uint4*)ldsU)[tid + r*THREADS] = ((const uint4*)Uw)[tid + r*THREADS];
#pragma unroll
  for (int r = 0; r < 2; ++r)
    ((uint4*)ldsG)[tid + r*THREADS] = ((const uint4*)Gw)[tid + r*THREADS];
  // zero own s0 buffer once; K=16..31 padding lanes stay zero forever
  uint4 z4 = {0, 0, 0, 0};
#pragma unroll
  for (int r = 0; r < 16; ++r)
    ((uint4*)ldsS0[wave])[lane + r*32] = z4;
  __syncthreads();

  const float bias0 = b_out[n16];
  const float bias1 = b_out[16 + n16];
  const int gw = blockIdx.x * WAVES_PER_BLOCK + wave;
  _Float16* myS0 = ldsS0[wave];

  for (int it = 0; it < TILES_PER_WAVE; ++it) {
    const int tile = gw * TILES_PER_WAVE + it;
    const int tokBase = tile * 16;
    if (it + 1 < TILES_PER_WAVE)   // pull next tile (2KB) toward L2/L0
      __builtin_prefetch(x + (size_t)(tokBase + 16) * EMBED + lane * 16, 0, 3);

    // ---- Phase A: product-state amplitudes s0 -> LDS (f16, K-major, padded)
#pragma unroll
    for (int p = 0; p < 4; ++p) {
      const int idx = p * 32 + lane;
      const int t = idx >> 3, h = idx & 7;       // lanes sweep heads fastest => coalesced
      const float4 a = *(const float4*)(x + (size_t)(tokBase + t) * EMBED + h * 4);
      float s0a, c0a, s1a, c1a, s2a, c2a, s3a, c3a;
      __sincosf(0.5f * a.x, &s0a, &c0a);
      __sincosf(0.5f * a.y, &s1a, &c1a);
      __sincosf(0.5f * a.z, &s2a, &c2a);
      __sincosf(0.5f * a.w, &s3a, &c3a);
      const float ab[4] = {c0a*c1a, c0a*s1a, s0a*c1a, s0a*s1a};  // qubits 0,1 (bits 3,2)
      const float cd[4] = {c2a*c3a, c2a*s3a, s2a*c3a, s2a*s3a};  // qubits 2,3 (bits 1,0)
      v8h lo, hiv;
#pragma unroll
      for (int k = 0; k < 8; ++k)  lo[k]      = (_Float16)(ab[k >> 2] * cd[k & 3]);
#pragma unroll
      for (int k = 8; k < 16; ++k) hiv[k - 8] = (_Float16)(ab[k >> 2] * cd[k & 3]);
      _Float16* dst = myS0 + (h*16 + t)*32;
      *(v8h*)dst       = lo;
      *(v8h*)(dst + 8) = hiv;
    }

    // ---- Phase B: per head, psi = U_h * s0^T via WMMA; P stays in registers
    v8h ph[8];
#pragma unroll
    for (int h = 0; h < 8; ++h) {
      const _Float16* ure = ldsU + ((h*2+0)*16 + n16)*32 + hi16*8;
      const _Float16* uim = ldsU + ((h*2+1)*16 + n16)*32 + hi16*8;
      v16h Are = ldpair(ure, 16);            // A: row=state(lane%16), K halves {0-7|8-15}+pad
      v16h Aim = ldpair(uim, 16);
      const _Float16* sb = myS0 + (h*16 + n16)*32 + hi16*8;
      v16h Bs = ldpair(sb, 16);              // B: col=token(lane%16), K halves {0-7|8-15}+pad
      v8f z8 = {};
      v8f dre = __builtin_amdgcn_wmma_f32_16x16x32_f16(false, Are, false, Bs, (short)0, z8, false, false);
      v8f dim = __builtin_amdgcn_wmma_f32_16x16x32_f16(false, Aim, false, Bs, (short)0, z8, false, false);
      v8f pr = dre * dre + dim * dim;        // |psi|^2, D layout [state row, token lane]
#pragma unroll
      for (int r = 0; r < 8; ++r) ph[h][r] = (_Float16)pr[r];
    }

    // ---- Phase C: out = P_all(16x128) @ G^T(128x32) + b ; P regs already in A layout
#pragma unroll
    for (int ntile = 0; ntile < 2; ++ntile) {
      v8f acc = {};
#pragma unroll
      for (int c = 0; c < 4; ++c) {          // K chunk = heads {2c, 2c+1}
        v16h A2 = __builtin_shufflevector(ph[2*c], ph[2*c+1],
                                          0,1,2,3,4,5,6,7,8,9,10,11,12,13,14,15);
        const _Float16* gb = ldsG + (ntile*16 + n16)*128 + c*32 + hi16*8;
        v16h Bg = ldpair(gb, 16);
        acc = __builtin_amdgcn_wmma_f32_16x16x32_f16(false, A2, false, Bg, (short)0, acc, false, false);
      }
      const float bias = ntile ? bias1 : bias0;
      float* obase = out + (size_t)(tokBase + hi16*8) * EMBED + ntile*16 + n16;
#pragma unroll
      for (int r = 0; r < 8; ++r)
        obase[(size_t)r * EMBED] = acc[r] + bias;
    }
  }
}

extern "C" void kernel_launch(void* const* d_in, const int* in_sizes, int n_in,
                              void* d_out, int out_size, void* d_ws, size_t ws_size,
                              hipStream_t stream) {
  (void)in_sizes; (void)n_in; (void)out_size; (void)ws_size;
  const float* x      = (const float*)d_in[0];
  const float* params = (const float*)d_in[1];
  const float* W_out  = (const float*)d_in[2];
  const float* b_out  = (const float*)d_in[3];
  _Float16* Uw = (_Float16*)d_ws;          // 8*2*16*32 halves = 16 KB
  _Float16* Gw = Uw + 8*2*16*32;           // 32*128 halves    =  8 KB
  qmha_setup<<<1, 128, 0, stream>>>(params, W_out, Uw, Gw);
  qmha_main<<<BLOCKS, THREADS, 0, stream>>>(x, b_out, Uw, Gw, (float*)d_out);
}
